// MultiheadAttention_70437463654944
// MI455X (gfx1250) — compile-verified, matrix operand order checked by reading, not run on hardware
//
#include <hip/hip_runtime.h>
#include <hip/hip_bf16.h>

typedef __bf16 bf16;
typedef __attribute__((ext_vector_type(16))) __bf16 v16bf;
typedef __attribute__((ext_vector_type(8)))  __bf16 v8bf;
typedef __attribute__((ext_vector_type(4)))  __bf16 v4bf;
typedef __attribute__((ext_vector_type(8)))  float  v8f;
typedef __attribute__((ext_vector_type(4)))  int    v4i;

#define WMMA_BF16(a, b, c) \
  __builtin_amdgcn_wmma_f32_16x16x32_bf16(false, (a), false, (b), (short)0, (c), false, false)

static constexpr int kB = 4, kS = 2048, kE = 1024, kH = 16, kD = 64;
static constexpr int kM = kB * kS;      // 8192 rows
static constexpr int kQKVN = 3 * kE;    // 3072
static constexpr int LDT = 56;          // padded LDS row stride (112B: conflict-free b128)
static constexpr int LDK = 72;          // padded K-tile stride (144B)

// ---------------------------------------------------------------------------
// Async global->LDS copy (16B per lane), tracked by ASYNCcnt.
// Builtin signature (probed via diagnostics): (AS1 v4i* src, AS3 v4i* dst,
// imm offset, imm cpol).
// ---------------------------------------------------------------------------
#if __has_builtin(__builtin_amdgcn_global_load_async_to_lds_b128)
__device__ __forceinline__ void async_copy_b128(const bf16* gsrc, bf16* ldst) {
  __builtin_amdgcn_global_load_async_to_lds_b128(
      (__attribute__((address_space(1))) v4i*)gsrc,
      (__attribute__((address_space(3))) v4i*)ldst, 0, 0);
}
#else
__device__ __forceinline__ void async_copy_b128(const bf16* gsrc, bf16* ldst) {
  unsigned lds_off = (unsigned)(unsigned long long)ldst;  // LDS aperture: addr[31:0]
  asm volatile("global_load_async_to_lds_b128 %0, %1, off"
               :: "v"(lds_off), "v"(gsrc) : "memory");
}
#endif

#if __has_builtin(__builtin_amdgcn_s_wait_asynccnt)
#define ASYNC_WAIT(n) __builtin_amdgcn_s_wait_asynccnt(n)
#else
#define ASYNC_WAIT(n) asm volatile("s_wait_asynccnt %0" :: "i"(n) : "memory")
#endif

// ---------------------------------------------------------------------------
// Fragment loader: 16x32 bf16 tile (row-major, leading dim `ld` elems) into
// the wave32 WMMA operand layout (ISA 7.12.2, 16-bit A 16x32):
//   lanes 0-15  : row = lane,    K in {0..7}  and {16..23}
//   lanes 16-31 : row = lane-16, K in {8..15} and {24..31}
// Same layout serves as B operand when the tile holds rows of B^T.
// ---------------------------------------------------------------------------
__device__ __forceinline__ v16bf load_frag(const bf16* __restrict__ base, int ld) {
  const int lane = threadIdx.x & 31;
  const int row  = lane & 15;
  const int kb   = (lane >> 4) << 3;          // 0 or 8
  const bf16* p  = base + row * ld + kb;
  union { v16bf v; v8bf h[2]; } u;
  u.h[0] = *(const v8bf*)(p);
  u.h[1] = *(const v8bf*)(p + 16);
  return u.v;
}

__device__ __forceinline__ v8f zero_v8f() {
  v8f z = {0.f, 0.f, 0.f, 0.f, 0.f, 0.f, 0.f, 0.f};
  return z;
}

// ---------------------------------------------------------------------------
// fp32 -> bf16 conversion (4 elements / thread, b128 loads)
// ---------------------------------------------------------------------------
__global__ __launch_bounds__(256) void cvt_f32_bf16(const float* __restrict__ src,
                                                    bf16* __restrict__ dst, int n4) {
  int i = blockIdx.x * 256 + threadIdx.x;
  if (i < n4) {
    float4 v = ((const float4*)src)[i];
    v4bf o;
    o[0] = (bf16)v.x; o[1] = (bf16)v.y; o[2] = (bf16)v.z; o[3] = (bf16)v.w;
    ((v4bf*)dst)[i] = o;
  }
}

// ---------------------------------------------------------------------------
// Shared GEMM core: C(128x256 block tile) = A[M,1024] @ B[N,1024]^T.
// Double-buffered async staging: per 32-K step, 2 async b128 ops stage the
// 128x32 A tile and 4 stage the 256x32 B tile (all 256 threads, 16B/lane).
// Each wave then computes a 64x64 tile from LDS (conflict-free padded rows).
// ---------------------------------------------------------------------------
__device__ __forceinline__ void gemm_staged(const bf16* __restrict__ Ag,
                                            const bf16* __restrict__ Bg,
                                            int m0b, int n0b,
                                            bf16* Asm, bf16* Bsm,
                                            v8f (&acc)[4][4]) {
  const int tid  = threadIdx.x;
  const int wave = tid >> 5;
  const int wm   = (wave >> 2) * 64;
  const int wn   = (wave & 3) * 64;

  auto issue = [&](int buf, int k0) {
    bf16* Ab = Asm + buf * (128 * LDT);
    bf16* Bb = Bsm + buf * (256 * LDT);
#pragma unroll
    for (int q = 0; q < 2; ++q) {               // 128 rows x 64B
      const int idx = q * 256 + tid, row = idx >> 2, ch = (idx & 3) * 8;
      async_copy_b128(Ag + (m0b + row) * kE + k0 + ch, Ab + row * LDT + ch);
    }
#pragma unroll
    for (int q = 0; q < 4; ++q) {               // 256 rows x 64B
      const int idx = q * 256 + tid, row = idx >> 2, ch = (idx & 3) * 8;
      async_copy_b128(Bg + (n0b + row) * kE + k0 + ch, Bb + row * LDT + ch);
    }
  };

  issue(0, 0);
  for (int k = 0; k < kE; k += 32) {
    const int buf = (k >> 5) & 1;
    if (k + 32 < kE) {
      issue(buf ^ 1, k + 32);
      ASYNC_WAIT(6);                            // drain previous group only
    } else {
      ASYNC_WAIT(0);
    }
    __syncthreads();                            // all waves' tiles landed

    const bf16* Ab = Asm + buf * (128 * LDT);
    const bf16* Bb = Bsm + buf * (256 * LDT);
    v16bf a[4], b[4];
#pragma unroll
    for (int i = 0; i < 4; ++i) a[i] = load_frag(Ab + (wm + 16 * i) * LDT, LDT);
#pragma unroll
    for (int j = 0; j < 4; ++j) b[j] = load_frag(Bb + (wn + 16 * j) * LDT, LDT);
#pragma unroll
    for (int i = 0; i < 4; ++i)
#pragma unroll
      for (int j = 0; j < 4; ++j) acc[i][j] = WMMA_BF16(a[i], b[j], acc[i][j]);

    __syncthreads();                            // protect buf from next prefetch
  }
}

// ---------------------------------------------------------------------------
// QKV projection: C[8192,3072] = Xb @ Wqkv^T + b_qkv, scattered to
//   c in [0,64)   -> Q [bh][s][64] * 1/sqrt(64)
//   c in [64,128) -> K [bh][s][64]
//   c in [128,192)-> Vt[bh][d][2048]  (transposed, packed b128 stores)
// ---------------------------------------------------------------------------
__global__ __launch_bounds__(256)
void gemm_qkv(const bf16* __restrict__ X, const bf16* __restrict__ W,
              const float* __restrict__ bias,
              bf16* __restrict__ Qd, bf16* __restrict__ Kd, bf16* __restrict__ Vt) {
  __shared__ bf16 Asm[2 * 128 * LDT];
  __shared__ bf16 Bsm[2 * 256 * LDT];
  const int lane = threadIdx.x & 31;
  const int wave = threadIdx.x >> 5;
  const int m0 = blockIdx.x * 128 + (wave >> 2) * 64;
  const int n0 = blockIdx.y * 256 + (wave & 3) * 64;

  v8f acc[4][4];
#pragma unroll
  for (int i = 0; i < 4; ++i)
#pragma unroll
    for (int j = 0; j < 4; ++j) acc[i][j] = zero_v8f();

  gemm_staged(X, W, blockIdx.x * 128, blockIdx.y * 256, Asm, Bsm, acc);

  const int nc = lane & 15;
  const int mb = (lane >> 4) << 3;
#pragma unroll
  for (int j = 0; j < 4; ++j) {
    const int n = n0 + 16 * j + nc;
    const int h = n / 192;
    const int c = n % 192;
    const float bv = bias[n];
#pragma unroll
    for (int i = 0; i < 4; ++i) {
      const int mrow = m0 + 16 * i + mb;        // 8-row run stays inside one batch
      const int bidx = mrow >> 11;
      const int s    = mrow & (kS - 1);
      const int bh   = bidx * kH + h;
      if (c < 128) {
        bf16* dst = (c < 64) ? Qd : Kd;
        const int   col = c & 63;
        const float sc  = (c < 64) ? 0.125f : 1.0f;
#pragma unroll
        for (int r = 0; r < 8; ++r)
          dst[((bh * kS + (s + r)) << 6) + col] = (bf16)((acc[i][j][r] + bv) * sc);
      } else {
        v8bf pk;
#pragma unroll
        for (int r = 0; r < 8; ++r) pk[r] = (bf16)(acc[i][j][r] + bv);
        *(v8bf*)(Vt + (bh * kD + (c - 128)) * kS + s) = pk;
      }
    }
  }
}

// ---------------------------------------------------------------------------
// Flash attention.  Grid = B*H*(S/128), 8 waves/block, 16 queries/wave.
// K/V tiles for each 32-key step are IDENTICAL for all 8 waves -> stage them
// once per block via async-to-LDS (2 ops/step), double-buffered.
//   S^T = K_tile x Q^T  (query index on lanes: softmax stats need only one
//   shfl_xor(16)); P bounces through per-wave LDS (C-layout f32 -> A-layout
//   bf16); O += P x V via B-frags read from the staged V^T tile.
// ---------------------------------------------------------------------------
__global__ __launch_bounds__(256)
void attn_flash(const bf16* __restrict__ Q, const bf16* __restrict__ K,
                const bf16* __restrict__ Vt, bf16* __restrict__ Vals) {
  __shared__ bf16 Ksm[2 * 32 * LDK];    // 32 keys x 64 feats (padded)
  __shared__ bf16 Vsm[2 * 64 * LDT];    // 64 dims x 32 keys  (padded)
  __shared__ bf16 Plds[8 * 16 * 32];    // per-wave P tile
  const int tid  = threadIdx.x;
  const int wave = tid >> 5;
  const int lane = tid & 31;
  const int ln   = lane & 15;
  const int mb   = (lane >> 4) << 3;

  const int bh = blockIdx.x >> 4;
  const int qt = blockIdx.x & 15;
  const int q0 = qt * 128 + wave * 16;

  const bf16* Qb = Q  + (bh * kS + q0) * kD;
  const bf16* Kb = K  + bh * kS * kD;
  const bf16* Vb = Vt + bh * kD * kS;
  bf16* pw = Plds + wave * 512;

  auto issue = [&](int buf, int kt) {
    bf16* Kt = Ksm + buf * (32 * LDK);
    bf16* Vd = Vsm + buf * (64 * LDT);
    {  // K tile: 32 rows x 128B
      const int row = tid >> 3, ch = (tid & 7) * 8;
      async_copy_b128(Kb + (kt + row) * kD + ch, Kt + row * LDK + ch);
    }
    {  // V^T tile: 64 rows x 64B
      const int row = tid >> 2, ch = (tid & 3) * 8;
      async_copy_b128(Vb + row * kS + kt + ch, Vd + row * LDT + ch);
    }
  };

  const v16bf qf0 = load_frag(Qb, kD);          // Q feat 0..31
  const v16bf qf1 = load_frag(Qb + 32, kD);     // Q feat 32..63

  v8f o[4];
#pragma unroll
  for (int d = 0; d < 4; ++d) o[d] = zero_v8f();
  float m_i = -1e30f, l_i = 0.f;

  issue(0, 0);
  for (int kt = 0; kt < kS; kt += 32) {
    const int buf = (kt >> 5) & 1;
    if (kt + 32 < kS) {
      issue(buf ^ 1, kt + 32);
      ASYNC_WAIT(2);
    } else {
      ASYNC_WAIT(0);
    }
    __syncthreads();

    const bf16* Kt = Ksm + buf * (32 * LDK);
    const bf16* Vd = Vsm + buf * (64 * LDT);

    v8f st[2];
#pragma unroll
    for (int j = 0; j < 2; ++j) {
      v16bf ka0 = load_frag(Kt + (16 * j) * LDK, LDK);
      v16bf ka1 = load_frag(Kt + (16 * j) * LDK + 32, LDK);
      st[j] = WMMA_BF16(ka0, qf0, zero_v8f());
      st[j] = WMMA_BF16(ka1, qf1, st[j]);       // S^T tile: m=key, n=query
    }

    // online softmax (lane's query = ln; key halves split across lane pair)
    float mx = -1e30f;
#pragma unroll
    for (int j = 0; j < 2; ++j)
#pragma unroll
      for (int r = 0; r < 8; ++r) mx = fmaxf(mx, st[j][r]);
    mx = fmaxf(mx, __shfl_xor(mx, 16, 32));
    const float mnew  = fmaxf(m_i, mx);
    const float alpha = __expf(m_i - mnew);
    m_i = mnew;

    float rsum = 0.f;
#pragma unroll
    for (int j = 0; j < 2; ++j)
#pragma unroll
      for (int r = 0; r < 8; ++r) {
        float p = __expf(st[j][r] - m_i);
        st[j][r] = p;
        rsum += p;
      }
    rsum += __shfl_xor(rsum, 16, 32);
    l_i = l_i * alpha + rsum;

    // P (C-layout f32) -> per-wave LDS row-major bf16 [query][key], 16B stores
#pragma unroll
    for (int j = 0; j < 2; ++j) {
      v8bf pk;
#pragma unroll
      for (int r = 0; r < 8; ++r) pk[r] = (bf16)st[j][r];
      *(v8bf*)(pw + ln * 32 + j * 16 + mb) = pk;
    }

    // rescale O accumulators by each row's alpha (broadcast from lane=query)
    float al[8];
#pragma unroll
    for (int r = 0; r < 8; ++r) al[r] = __shfl(alpha, mb + r, 32);
#pragma unroll
    for (int d = 0; d < 4; ++d)
#pragma unroll
      for (int r = 0; r < 8; ++r) o[d][r] *= al[r];

    // O(16x64) += P(16x32) x V(32x64)
    v16bf pa = load_frag(pw, 32);
#pragma unroll
    for (int d = 0; d < 4; ++d) {
      v16bf vfr = load_frag(Vd + (16 * d) * LDT, LDT);
      o[d] = WMMA_BF16(pa, vfr, o[d]);
    }
    __syncthreads();
  }

  // finalize: O /= l, store bf16 to Vals[b*S+q][h*64+d]
  float linv[8];
#pragma unroll
  for (int r = 0; r < 8; ++r) linv[r] = 1.f / __shfl(l_i, mb + r, 32);
  const int b = bh >> 4, h = bh & 15;
#pragma unroll
  for (int d = 0; d < 4; ++d)
#pragma unroll
    for (int r = 0; r < 8; ++r) {
      const int qrow = q0 + mb + r;
      Vals[(size_t)(b * kS + qrow) * kE + h * kD + d * 16 + ln] =
          (bf16)(o[d][r] * linv[r]);
    }
}

// ---------------------------------------------------------------------------
// Output projection: Out[8192,1024] = Vals @ Wout^T + b_out (f32 output)
// ---------------------------------------------------------------------------
__global__ __launch_bounds__(256)
void gemm_out(const bf16* __restrict__ Vl, const bf16* __restrict__ W,
              const float* __restrict__ bias, float* __restrict__ Out) {
  __shared__ bf16 Asm[2 * 128 * LDT];
  __shared__ bf16 Bsm[2 * 256 * LDT];
  const int lane = threadIdx.x & 31;
  const int wave = threadIdx.x >> 5;
  const int m0 = blockIdx.x * 128 + (wave >> 2) * 64;
  const int n0 = blockIdx.y * 256 + (wave & 3) * 64;

  v8f acc[4][4];
#pragma unroll
  for (int i = 0; i < 4; ++i)
#pragma unroll
    for (int j = 0; j < 4; ++j) acc[i][j] = zero_v8f();

  gemm_staged(Vl, W, blockIdx.x * 128, blockIdx.y * 256, Asm, Bsm, acc);

  const int nc = lane & 15;
  const int mb = (lane >> 4) << 3;
#pragma unroll
  for (int j = 0; j < 4; ++j) {
    const int n = n0 + 16 * j + nc;
    const float bv = bias[n];
#pragma unroll
    for (int i = 0; i < 4; ++i) {
      const int mrow = m0 + 16 * i + mb;
#pragma unroll
      for (int r = 0; r < 8; ++r)
        Out[(size_t)(mrow + r) * kE + n] = acc[i][j][r] + bv;
    }
  }
}

// ---------------------------------------------------------------------------
extern "C" void kernel_launch(void* const* d_in, const int* in_sizes, int n_in,
                              void* d_out, int out_size, void* d_ws, size_t ws_size,
                              hipStream_t stream) {
  (void)in_sizes; (void)n_in; (void)out_size; (void)ws_size;
  const float* x     = (const float*)d_in[0];   // [4,1,2048,1024]
  const float* w_qkv = (const float*)d_in[1];   // [3072,1024]
  const float* b_qkv = (const float*)d_in[2];   // [3072]
  const float* w_out = (const float*)d_in[3];   // [1024,1024]
  const float* b_out = (const float*)d_in[4];   // [1024]
  float* out = (float*)d_out;                   // [4,2048,1024]

  char* ws = (char*)d_ws;
  bf16* Xb   = (bf16*)ws;  ws += (size_t)kM * kE * 2;           // 16 MB
  bf16* Wqb  = (bf16*)ws;  ws += (size_t)kQKVN * kE * 2;        //  6 MB
  bf16* Wob  = (bf16*)ws;  ws += (size_t)kE * kE * 2;           //  2 MB
  bf16* Qd   = (bf16*)ws;  ws += (size_t)kB * kH * kS * kD * 2; // 16 MB
  bf16* Kd   = (bf16*)ws;  ws += (size_t)kB * kH * kS * kD * 2; // 16 MB
  bf16* Vt   = (bf16*)ws;  ws += (size_t)kB * kH * kS * kD * 2; // 16 MB
  bf16* Vals = (bf16*)ws;                                       // 16 MB

  {
    int n4 = kM * kE / 4;
    cvt_f32_bf16<<<(n4 + 255) / 256, 256, 0, stream>>>(x, Xb, n4);
    n4 = kQKVN * kE / 4;
    cvt_f32_bf16<<<(n4 + 255) / 256, 256, 0, stream>>>(w_qkv, Wqb, n4);
    n4 = kE * kE / 4;
    cvt_f32_bf16<<<(n4 + 255) / 256, 256, 0, stream>>>(w_out, Wob, n4);
  }
  gemm_qkv<<<dim3(kM / 128, kQKVN / 256), 256, 0, stream>>>(Xb, Wqb, b_qkv, Qd, Kd, Vt);
  attn_flash<<<kB * kH * (kS / 128), 256, 0, stream>>>(Qd, Kd, Vt, Vals);
  gemm_out<<<dim3(kM / 128, kE / 256), 256, 0, stream>>>(Vals, Wob, b_out, out);
}
